// cross_attention_32925219291320
// MI455X (gfx1250) — matrix operand order repaired and checked by reading, not compiled, not run on hardware
//
#include <hip/hip_runtime.h>
#include <hip/hip_bf16.h>

typedef __bf16 v16bf __attribute__((ext_vector_type(16)));
typedef __bf16 v8bf  __attribute__((ext_vector_type(8)));
typedef __bf16 v4bf  __attribute__((ext_vector_type(4)));
typedef float  v8f   __attribute__((ext_vector_type(8)));
typedef float  v4f   __attribute__((ext_vector_type(4)));

#define TOKENS 1024
#define DMODEL 256
#define CHUNK  16
#define NCHUNK (TOKENS / CHUNK)
#define WSTR   264   // bf16 LDS row stride: 528B rows -> 16B aligned, 4-bank row skew
#define XSTR   260   // fp32 LDS row stride: 1040B rows -> 16B aligned, 4-bank row skew

// CDNA5 async copy global -> LDS (per-lane), tracked by ASYNCcnt (ISA 08 §4.4)
__device__ __forceinline__ void async_ld_b128(unsigned lds_off, const float* g) {
  asm volatile("global_load_async_to_lds_b128 %0, %1, off"
               :: "v"(lds_off), "v"((unsigned long long)(uintptr_t)g) : "memory");
}
__device__ __forceinline__ void wait_async0() {
  asm volatile("s_wait_asynccnt 0" ::: "memory");
}

__global__ __launch_bounds__(256) void fused_cross_attn(
    const float* __restrict__ x,
    const float* __restrict__ wq, const float* __restrict__ bq,
    const float* __restrict__ wk, const float* __restrict__ bk,
    const float* __restrict__ wv, const float* __restrict__ bv,
    const float* __restrict__ wo, const float* __restrict__ bo,
    float* __restrict__ out)
{
  // 270336 (wK,wV) + 33280 (xf x2) + 16896 (ks,vs) + 3072 + 512 = 324,096 B
  __shared__ __align__(16) float  xf[2][CHUNK * XSTR];
  __shared__ __align__(16) __bf16 wK[DMODEL * WSTR];
  __shared__ __align__(16) __bf16 wV[DMODEL * WSTR];
  __shared__ __align__(16) __bf16 ks[CHUNK * WSTR];
  __shared__ __align__(16) __bf16 vs[CHUNK * WSTR];
  __shared__ float  qs[DMODEL];
  __shared__ float  x0s[DMODEL];
  __shared__ float  lg[128];          // 8 heads x 16 tokens
  __shared__ float  att[DMODEL];

  const int tid  = threadIdx.x;
  const int wgid = blockIdx.x;                        // b*S + s
  const float* xbase = x + (size_t)wgid * TOKENS * DMODEL;

  // ---- fire async copy of chunk 0 into xf[0] (overlaps weight staging) ----
  #pragma unroll
  for (int sweep = 0; sweep < 4; ++sweep) {
    int i = tid * 4 + sweep * 1024;                   // float index in chunk
    int row = i >> 8, col = i & 255;
    async_ld_b128((unsigned)(uintptr_t)&xf[0][row * XSTR + col], xbase + i);
  }

  // ---- stage K & V weight matrices into LDS as bf16 (row-major, padded) ----
  for (int i = tid * 4; i < DMODEL * DMODEL; i += 1024) {
    int row = i >> 8, col = i & 255;
    float4 fk = *(const float4*)(wk + i);
    float4 fv = *(const float4*)(wv + i);
    v4bf bk4 = { (__bf16)fk.x, (__bf16)fk.y, (__bf16)fk.z, (__bf16)fk.w };
    v4bf bv4 = { (__bf16)fv.x, (__bf16)fv.y, (__bf16)fv.z, (__bf16)fv.w };
    *(v4bf*)(wK + row * WSTR + col) = bk4;
    *(v4bf*)(wV + row * WSTR + col) = bv4;
  }
  x0s[tid] = xbase[tid];                              // token 0 row (query source)

  // ---- per-wave WMMA role: 32 tiles/chunk = 2 mat x 16 ntile, 4 tiles/wave ----
  const int wave = tid >> 5;                          // wave32
  const int lane = tid & 31;
  const int ln   = lane & 15, lhi = lane >> 4;
  const int mat  = wave >> 2;                         // 0 = K, 1 = V
  const int nt0  = (wave & 3) * 4;                    // 4 consecutive N-tiles
  const __bf16* wMat = mat ? wV : wK;
  const float*  bias = mat ? bv : bk;
  __bf16*       kvout = mat ? vs : ks;

  // hoist loop-invariant bias lanes into registers: the "memory" clobber on the
  // async asm would otherwise force a global reload + loadcnt wait every chunk
  float bcol[4];
  #pragma unroll
  for (int t = 0; t < 4; ++t) bcol[t] = bias[16 * (nt0 + t) + ln];

  wait_async0();
  __syncthreads();

  // ---- q projection in fp32 (tiny), fold in 1/sqrt(dh) ----
  {
    const float* wr = wq + tid * DMODEL;              // y_j = sum_i x_i * W[j,i]
    float a = bq[tid];
    #pragma unroll 8
    for (int i = 0; i < DMODEL; ++i) a += x0s[i] * wr[i];
    qs[tid] = a * 0.17677669529663687f;               // 1/sqrt(32)
  }

  // ---- online-softmax state: thread owns (head = tid>>5, dim = tid&31) ----
  const int h = tid >> 5, dd = tid & 31;
  float m = -INFINITY, l = 0.f, oacc = 0.f;

  for (int ch = 0; ch < NCHUNK; ++ch) {
    const int cur = ch & 1;
    // fire async copy of next chunk into the other buffer (hidden behind compute)
    if (ch + 1 < NCHUNK) {
      const float* xc = xbase + (size_t)(ch + 1) * CHUNK * DMODEL;
      #pragma unroll
      for (int sweep = 0; sweep < 4; ++sweep) {
        int i = tid * 4 + sweep * 1024;
        int row = i >> 8, col = i & 255;
        async_ld_b128((unsigned)(uintptr_t)&xf[cur ^ 1][row * XSTR + col], xc + i);
      }
    }

    // A fragments (16x32 bf16) built from fp32 LDS: cols {32kk+8hi+0..7, +16..23}
    const float* xcur = xf[cur];
    v16bf A[8];
    #pragma unroll
    for (int kk = 0; kk < 8; ++kk) {
      const float* ap = xcur + ln * XSTR + 32 * kk + 8 * lhi;
      v4f a0 = *(const v4f*)(ap);
      v4f a1 = *(const v4f*)(ap + 4);
      v4f a2 = *(const v4f*)(ap + 16);
      v4f a3 = *(const v4f*)(ap + 20);
      v16bf t;
      t[0]  = (__bf16)a0[0]; t[1]  = (__bf16)a0[1]; t[2]  = (__bf16)a0[2]; t[3]  = (__bf16)a0[3];
      t[4]  = (__bf16)a1[0]; t[5]  = (__bf16)a1[1]; t[6]  = (__bf16)a1[2]; t[7]  = (__bf16)a1[3];
      t[8]  = (__bf16)a2[0]; t[9]  = (__bf16)a2[1]; t[10] = (__bf16)a2[2]; t[11] = (__bf16)a2[3];
      t[12] = (__bf16)a3[0]; t[13] = (__bf16)a3[1]; t[14] = (__bf16)a3[2]; t[15] = (__bf16)a3[3];
      A[kk] = t;
    }
    // K/V projection tiles: C init = bias, 8 WMMAs over K=256
    #pragma unroll
    for (int t = 0; t < 4; ++t) {
      int nt = nt0 + t;
      float bc = bcol[t];
      v8f c = { bc, bc, bc, bc, bc, bc, bc, bc };
      #pragma unroll
      for (int kk = 0; kk < 8; ++kk) {
        // B (32x16) of W^T == bf16 cols {32kk+8hi+0..7, +16..23} of row-major W per lane
        const __bf16* bp = wMat + (16 * nt + ln) * WSTR + 32 * kk + 8 * lhi;
        v8bf blo = *(const v8bf*)bp;
        v8bf bhi = *(const v8bf*)(bp + 16);
        v16bf B = __builtin_shufflevector(blo, bhi, 0,1,2,3,4,5,6,7,8,9,10,11,12,13,14,15);
        c = __builtin_amdgcn_wmma_f32_16x16x32_bf16(false, A[kk], false, B,
                                                    (short)0, c, false, false);
      }
      #pragma unroll
      for (int r = 0; r < 8; ++r)   // C layout: reg r -> row r+8*lhi, col ln
        kvout[(r + 8 * lhi) * WSTR + 16 * nt + ln] = (__bf16)c[r];
    }
    __syncthreads();

    // logits: one (head, token) pair per thread (waves 0-3)
    if (tid < 128) {
      const int hh = tid >> 4, n = tid & 15;
      const __bf16* kr = ks + n * WSTR + hh * 32;
      const float*  qh = qs + hh * 32;
      float s = 0.f;
      #pragma unroll 8
      for (int i = 0; i < 32; ++i) s += qh[i] * (float)kr[i];
      lg[tid] = s;                                    // lg[hh*16+n]
    }
    __syncthreads();

    // online softmax + PV accumulate
    {
      const float* lgh = lg + h * 16;
      float cm = -INFINITY;
      #pragma unroll 8
      for (int n = 0; n < 16; ++n) cm = fmaxf(cm, lgh[n]);
      float mn = fmaxf(m, cm);
      float sc = __expf(m - mn);
      oacc *= sc; l *= sc; m = mn;
      #pragma unroll 4
      for (int n = 0; n < 16; ++n) {
        float p = __expf(lgh[n] - m);
        l += p;
        oacc += p * (float)vs[n * WSTR + h * 32 + dd];
      }
    }
    wait_async0();       // own async copies into xf[next] complete
    __syncthreads();     // all waves' copies complete; safe to read next buffer
  }

  att[tid] = oacc / l;   // tid == h*32 + dd
  __syncthreads();

  // ---- output projection (fp32, wo stays in L2) ----
  {
    const float* wr = wo + tid * DMODEL;
    float o = bo[tid];
    #pragma unroll 8
    for (int i = 0; i < DMODEL; ++i) o += att[i] * wr[i];
    out[(size_t)wgid * DMODEL + tid] = o;
  }
}

extern "C" void kernel_launch(void* const* d_in, const int* in_sizes, int n_in,
                              void* d_out, int out_size, void* d_ws, size_t ws_size,
                              hipStream_t stream) {
  const float* x  = (const float*)d_in[0];
  const float* wq = (const float*)d_in[1];
  const float* bq = (const float*)d_in[2];
  const float* wk = (const float*)d_in[3];
  const float* bk = (const float*)d_in[4];
  const float* wv = (const float*)d_in[5];
  const float* bv = (const float*)d_in[6];
  const float* wo = (const float*)d_in[7];
  const float* bo = (const float*)d_in[8];
  float* out = (float*)d_out;

  const int n_bs = in_sizes[0] / (TOKENS * DMODEL);   // B*S = 512
  fused_cross_attn<<<n_bs, 256, 0, stream>>>(x, wq, bq, wk, bk, wv, bv, wo, bo, out);
}
